// Model_69612829934231
// MI455X (gfx1250) — compile-verified, matrix operand order checked by reading, not run on hardware
//
#include <hip/hip_runtime.h>

#define NHEAD 4
#define NEG_SLOPE 0.2f

typedef __attribute__((ext_vector_type(16))) __bf16 v16bf;
typedef __attribute__((ext_vector_type(8)))  __bf16 v8bf;
typedef __attribute__((ext_vector_type(8)))  float  v8f;

// ---------------------------------------------------------------- utilities
__global__ void fill_f32(float* p, float v, long long n) {
    long long i = (long long)blockIdx.x * blockDim.x + threadIdx.x;
    long long st = (long long)gridDim.x * blockDim.x;
    for (; i < n; i += st) p[i] = v;
}

__global__ void fill_u32(unsigned* p, unsigned v, long long n) {
    long long i = (long long)blockIdx.x * blockDim.x + threadIdx.x;
    long long st = (long long)gridDim.x * blockDim.x;
    for (; i < n; i += st) p[i] = v;
}

__device__ __forceinline__ unsigned f32_ord(float f) {
    unsigned u = __float_as_uint(f);
    return (u >> 31) ? ~u : (u | 0x80000000u);
}
__device__ __forceinline__ float ord_f32(unsigned v) {
    return (v >> 31) ? __uint_as_float(v & 0x7FFFFFFFu) : __uint_as_float(~v);
}
#define ORD_NEG_INF 0x007FFFFFu   // f32_ord(-inf)

// ---------------------------------------------------------- operand packing
// A[M,K] f32 row-major -> Apack[M,Kpad] bf16 row-major, zero-padded K tail.
__global__ void pack_A(const float* __restrict__ A, __bf16* __restrict__ Apack,
                       long long M, int K, int Kpad)
{
    long long i = (long long)blockIdx.x * blockDim.x + threadIdx.x;
    long long tot = M * Kpad;
    long long st = (long long)gridDim.x * blockDim.x;
    for (; i < tot; i += st) {
        long long row = i / Kpad; int kp = (int)(i % Kpad);
        Apack[i] = (__bf16)((kp < K) ? A[row * K + kp] : 0.f);
    }
}

// B[K,Nc] f32 -> WMMA B-fragment order: Bpack[((ks*Nc + col)*2 + half)*16 + j]
// half 0 owns k = ks*32 + {0..7,16..23}; half 1 owns k = ks*32 + {8..15,24..31}
__global__ void pack_B(const float* __restrict__ B, __bf16* __restrict__ Bpack,
                       int K, int Kpad, int Nc)
{
    long long i = (long long)blockIdx.x * blockDim.x + threadIdx.x;
    long long tot = (long long)Kpad * Nc;
    long long st = (long long)gridDim.x * blockDim.x;
    for (; i < tot; i += st) {
        int j    = (int)(i & 15);
        int half = (int)((i >> 4) & 1);
        long long cn = i >> 5;            // ks*Nc + col
        int col = (int)(cn % Nc);
        int ks  = (int)(cn / Nc);
        int k = ks * 32 + (j & 7) + ((j >> 3) << 4) + (half << 3);
        Bpack[i] = (__bf16)((k < K) ? B[(long long)k * Nc + col] : 0.f);
    }
}

// ------------------------------------------------------- bf16 WMMA GEMM
// C[M,Nc] = A@B (+bias) (+add[idx[row]*Nc+col]) (optional relu).
// Requires M%16==0, Nc%64==0, Kpad%32==0 (guaranteed by construction).
// One wave per 16x64 tile: per k-step, 1 A fragment is reused across 4
// independent v_wmma_f32_16x16x32_bf16 (hides WMMA latency, cuts A traffic 4x).
__global__ void gemm16_bf16(const __bf16* __restrict__ Apack, const __bf16* __restrict__ Bpack,
                            const float* __restrict__ bias,
                            const float* __restrict__ add, const int* __restrict__ idx,
                            float* __restrict__ C,
                            int M, int Kpad, int Nc, int relu)
{
    int wave = (int)((blockIdx.x * (long long)blockDim.x + threadIdx.x) >> 5);
    int lane = threadIdx.x & 31;
    int Mt = M >> 4, Ng = Nc >> 6;             // 64-wide column groups
    if (wave >= Mt * Ng) return;               // uniform per wave: EXEC stays full
    int tm = wave / Ng, tg = wave % Ng;
    int half = lane >> 4, l = lane & 15;

    const __bf16* aptr = Apack + ((long long)(tm * 16 + l)) * Kpad + half * 8;
    // fragment base for sub-tile 0; sub-tiles step by 16 columns = 16*32 bf16
    const __bf16* bptr = Bpack + ((long long)(tg * 64 + l) * 2 + half) * 16;
    long long bstride = (long long)Nc * 32;    // per k-step

    v8f acc0 = {0.f,0.f,0.f,0.f,0.f,0.f,0.f,0.f};
    v8f acc1 = acc0, acc2 = acc0, acc3 = acc0;
    for (int k0 = 0; k0 < Kpad; k0 += 32) {
        v8bf a0 = *(const v8bf*)(aptr);        // k = k0 + half*8 + {0..7}
        v8bf a1 = *(const v8bf*)(aptr + 16);   // k = k0 + 16 + half*8 + {0..7}
        v16bf a = __builtin_shufflevector(a0, a1, 0,1,2,3,4,5,6,7,8,9,10,11,12,13,14,15);
        v16bf b0 = *(const v16bf*)(bptr);
        v16bf b1 = *(const v16bf*)(bptr + 16 * 32);
        v16bf b2 = *(const v16bf*)(bptr + 32 * 32);
        v16bf b3 = *(const v16bf*)(bptr + 48 * 32);
        acc0 = __builtin_amdgcn_wmma_f32_16x16x32_bf16(false, a, false, b0, (short)0, acc0, false, false);
        acc1 = __builtin_amdgcn_wmma_f32_16x16x32_bf16(false, a, false, b1, (short)0, acc1, false, false);
        acc2 = __builtin_amdgcn_wmma_f32_16x16x32_bf16(false, a, false, b2, (short)0, acc2, false, false);
        acc3 = __builtin_amdgcn_wmma_f32_16x16x32_bf16(false, a, false, b3, (short)0, acc3, false, false);
        aptr += 32;
        bptr += bstride;
    }

    v8f accs[4] = {acc0, acc1, acc2, acc3};
#pragma unroll
    for (int sub = 0; sub < 4; ++sub) {
        int col = tg * 64 + sub * 16 + l;
        float bi = bias ? bias[col] : 0.f;
#pragma unroll
        for (int r = 0; r < 8; ++r) {
            int row = tm * 16 + r + half * 8;  // C layout: vgpr r -> M = r + half*8
            float v = accs[sub][r] + bi;
            if (add) v += add[(long long)(idx ? idx[row] : row) * Nc + col];
            if (relu) v = v > 0.f ? v : 0.f;
            C[(long long)row * Nc + col] = v;
        }
    }
}

// ------------------------------------------------ attention helper kernels
// weff[k,h] = sum_c W[k, h*C+c] * a[h,c]   (folds hd-projection into a Kin x 4 mat)
__global__ void make_weff(const float* __restrict__ W, const float* __restrict__ a,
                          float* __restrict__ weff, int Kin, int C)
{
    int i = blockIdx.x * blockDim.x + threadIdx.x;
    if (i >= Kin * NHEAD) return;
    int k = i / NHEAD, h = i % NHEAD;
    float s = 0.f;
    for (int c = 0; c < C; ++c) s += W[(long long)k * (NHEAD * C) + h * C + c] * a[h * C + c];
    weff[k * NHEAD + h] = s;
}

// out[n,h] = sum_k x[n,k] * weff[k,h]
__global__ void attn_logits_node(const float* __restrict__ x, const float* __restrict__ weff,
                                 float* __restrict__ out, long long Nn, int Kin)
{
    long long i = (long long)blockIdx.x * blockDim.x + threadIdx.x;
    long long tot = Nn * NHEAD;
    long long st = (long long)gridDim.x * blockDim.x;
    for (; i < tot; i += st) {
        long long n = i / NHEAD; int h = (int)(i % NHEAD);
        float s = 0.f;
        for (int k = 0; k < Kin; ++k) s += x[n * Kin + k] * weff[k * NHEAD + h];
        out[i] = s;
    }
}

__global__ void edge_logits(const int* __restrict__ src, const int* __restrict__ dst,
                            const float* __restrict__ es, const float* __restrict__ ed,
                            float* __restrict__ ebuf, unsigned* __restrict__ menc, long long En)
{
    long long i = (long long)blockIdx.x * blockDim.x + threadIdx.x;
    long long tot = En * NHEAD;
    long long st = (long long)gridDim.x * blockDim.x;
    for (; i < tot; i += st) {
        long long e = i / NHEAD; int h = (int)(i % NHEAD);
        float v = es[(long long)src[e] * NHEAD + h] + ed[(long long)dst[e] * NHEAD + h];
        v = v > 0.f ? v : NEG_SLOPE * v;       // leaky_relu
        ebuf[i] = v;
        atomicMax(&menc[(long long)dst[e] * NHEAD + h], f32_ord(v));
    }
}

__global__ void decode_max(unsigned* __restrict__ menc, long long n)
{
    long long i = (long long)blockIdx.x * blockDim.x + threadIdx.x;
    long long st = (long long)gridDim.x * blockDim.x;
    float* m = (float*)menc;
    for (; i < n; i += st) {
        unsigned v = menc[i];
        m[i] = (v == ORD_NEG_INF) ? 0.f : ord_f32(v);  // isolated dst -> 0
    }
}

__global__ void edge_exp(const int* __restrict__ dst, const float* __restrict__ m,
                         float* __restrict__ ebuf, float* __restrict__ den, long long En)
{
    long long i = (long long)blockIdx.x * blockDim.x + threadIdx.x;
    long long tot = En * NHEAD;
    long long st = (long long)gridDim.x * blockDim.x;
    for (; i < tot; i += st) {
        long long e = i / NHEAD; int h = (int)(i % NHEAD);
        float ex = __expf(ebuf[i] - m[(long long)dst[e] * NHEAD + h]);
        ebuf[i] = ex;
        atomicAdd(&den[(long long)dst[e] * NHEAD + h], ex);
    }
}

__global__ void edge_alpha(const int* __restrict__ dst, const float* __restrict__ den,
                           float* __restrict__ ebuf, long long En)
{
    long long i = (long long)blockIdx.x * blockDim.x + threadIdx.x;
    long long tot = En * NHEAD;
    long long st = (long long)gridDim.x * blockDim.x;
    for (; i < tot; i += st) {
        long long e = i / NHEAD; int h = (int)(i % NHEAD);
        ebuf[i] = ebuf[i] / (den[(long long)dst[e] * NHEAD + h] + 1e-16f);
    }
}

// one block per edge: coalesced message gather + float atomic scatter-add
__global__ void edge_scatter(const int* __restrict__ src, const int* __restrict__ dst,
                             const float* __restrict__ hs, const float* __restrict__ alpha,
                             float* __restrict__ agg, int HC, int C)
{
    long long e = blockIdx.x;
    int s = src[e], d = dst[e];
    for (int ch = threadIdx.x; ch < HC; ch += blockDim.x) {
        int h = ch / C;
        atomicAdd(&agg[(long long)d * HC + ch],
                  hs[(long long)s * HC + ch] * alpha[e * NHEAD + h]);
    }
}

// x_out[n,c] = mean_h relu(agg[n,h*C+c] + b[h*C+c])
__global__ void head_mean(const float* __restrict__ agg, const float* __restrict__ b,
                          float* __restrict__ xout, long long Nn, int C)
{
    long long i = (long long)blockIdx.x * blockDim.x + threadIdx.x;
    long long tot = Nn * C;
    long long st = (long long)gridDim.x * blockDim.x;
    for (; i < tot; i += st) {
        long long n = i / C; int c = (int)(i % C);
        float s = 0.f;
        for (int h = 0; h < NHEAD; ++h) {
            float v = agg[n * (long long)(NHEAD * C) + h * C + c] + b[h * C + c];
            s += v > 0.f ? v : 0.f;
        }
        xout[i] = s * (1.0f / NHEAD);
    }
}

// x[n,:] *= exp(x[n,:] @ W + b)   (C = 32)
__global__ void penalty_scale(float* __restrict__ x, const float* __restrict__ W,
                              const float* __restrict__ b, long long Nn)
{
    long long n = (long long)blockIdx.x * blockDim.x + threadIdx.x;
    if (n >= Nn) return;
    float s = b[0];
#pragma unroll
    for (int c = 0; c < 32; ++c) s += x[n * 32 + c] * W[c];
    float sc = __expf(s);
#pragma unroll
    for (int c = 0; c < 32; ++c) x[n * 32 + c] *= sc;
}

__global__ void gather_ef(const float* __restrict__ xd, const float* __restrict__ xl,
                          const int* __restrict__ e0, const int* __restrict__ e1,
                          float* __restrict__ ef, long long ELn)
{
    long long i = (long long)blockIdx.x * blockDim.x + threadIdx.x;
    long long tot = ELn * 64;
    long long st = (long long)gridDim.x * blockDim.x;
    for (; i < tot; i += st) {
        long long e = i >> 6; int c = (int)(i & 63);
        ef[i] = (c < 32) ? xd[(long long)e0[e] * 32 + c]
                         : xl[(long long)e1[e] * 32 + (c - 32)];
    }
}

__global__ void fc2_dot(const float* __restrict__ h, const float* __restrict__ W,
                        const float* __restrict__ b, float* __restrict__ out, long long ELn)
{
    long long e = (long long)blockIdx.x * blockDim.x + threadIdx.x;
    if (e >= ELn) return;
    float s = b[0];
#pragma unroll
    for (int c = 0; c < 64; ++c) s += h[e * 64 + c] * W[c];
    out[e] = s;
}

// ---------------------------------------------------------------- host side
static inline int nblk(long long n, int t) {
    long long b = (n + t - 1) / t;
    if (b > (1 << 26)) b = 1 << 26;
    if (b < 1) b = 1;
    return (int)b;
}

// full GEMM pipeline: pack A (f32->bf16, K padded), pack B (fragment order), WMMA
static void gemm_full(const float* A, const float* B, const float* bias,
                      const float* add, const int* idx, float* C,
                      int M, int K, int Nc, int relu,
                      __bf16* Apack, __bf16* Bpack, hipStream_t s)
{
    int Kpad = (K + 31) & ~31;
    pack_A<<<nblk((long long)M * Kpad, 256), 256, 0, s>>>(A, Apack, M, K, Kpad);
    pack_B<<<nblk((long long)Kpad * Nc, 256), 256, 0, s>>>(B, Bpack, K, Kpad, Nc);
    long long waves = (long long)(M / 16) * (Nc / 64);
    gemm16_bf16<<<(int)((waves * 32 + 255) / 256), 256, 0, s>>>(
        Apack, Bpack, bias, add, idx, C, M, Kpad, Nc, relu);
}

// one GAT direction: agg over edges src->dst, then head-mean into xout[Nn,C]
static void run_gat(const float* xsrc, const float* xdst,
                    const float* Ws, const float* Wd, const float* as_, const float* ad_,
                    const float* bgat, const int* src, const int* dst,
                    int Kin, int C, float* xout,
                    float* hs, float* agg, float* es, float* ed,
                    unsigned* menc, float* den, float* ebuf,
                    float* weff_s, float* weff_d,
                    __bf16* Apack, __bf16* Bpack,
                    long long Nn, long long En, hipStream_t s)
{
    int HC = NHEAD * C;
    make_weff<<<nblk(Kin * NHEAD, 128), 128, 0, s>>>(Ws, as_, weff_s, Kin, C);
    make_weff<<<nblk(Kin * NHEAD, 128), 128, 0, s>>>(Wd, ad_, weff_d, Kin, C);
    attn_logits_node<<<nblk(Nn * NHEAD, 256), 256, 0, s>>>(xsrc, weff_s, es, Nn, Kin);
    attn_logits_node<<<nblk(Nn * NHEAD, 256), 256, 0, s>>>(xdst, weff_d, ed, Nn, Kin);
    gemm_full(xsrc, Ws, nullptr, nullptr, nullptr, hs, (int)Nn, Kin, HC, 0, Apack, Bpack, s);

    fill_u32<<<nblk(Nn * NHEAD, 256), 256, 0, s>>>(menc, ORD_NEG_INF, Nn * NHEAD);
    fill_f32<<<nblk(Nn * NHEAD, 256), 256, 0, s>>>(den, 0.f, Nn * NHEAD);
    fill_f32<<<nblk(Nn * (long long)HC, 256), 256, 0, s>>>(agg, 0.f, Nn * (long long)HC);

    edge_logits<<<nblk(En * NHEAD, 256), 256, 0, s>>>(src, dst, es, ed, ebuf, menc, En);
    decode_max<<<nblk(Nn * NHEAD, 256), 256, 0, s>>>(menc, Nn * NHEAD);
    edge_exp<<<nblk(En * NHEAD, 256), 256, 0, s>>>(dst, (const float*)menc, ebuf, den, En);
    edge_alpha<<<nblk(En * NHEAD, 256), 256, 0, s>>>(dst, den, ebuf, En);
    edge_scatter<<<(int)En, 128, 0, s>>>(src, dst, hs, ebuf, agg, HC, C);

    head_mean<<<nblk(Nn * C, 256), 256, 0, s>>>(agg, bgat, xout, Nn, C);
}

extern "C" void kernel_launch(void* const* d_in, const int* in_sizes, int n_in,
                              void* d_out, int out_size, void* d_ws, size_t ws_size,
                              hipStream_t stream)
{
    const long long N = 50000, E = 300000, EL = 200000;

    const float* x_d    = (const float*)d_in[0];
    const float* x_l    = (const float*)d_in[1];
    const float* Wd_lin = (const float*)d_in[2];
    const float* bd_lin = (const float*)d_in[3];
    const float* Wl_lin = (const float*)d_in[4];
    const float* bl_lin = (const float*)d_in[5];
    const float* emb_d  = (const float*)d_in[6];
    const float* emb_l  = (const float*)d_in[7];
    const float* g[3][2][5];   // [layer][d2l/l2d][Ws,Wd,as,ad,b] -> d_in 8..37
    int base = 8;
    for (int L = 0; L < 3; ++L)
        for (int t = 0; t < 2; ++t)
            for (int p = 0; p < 5; ++p)
                g[L][t][p] = (const float*)d_in[base++];
    const float* pen_Wd = (const float*)d_in[38];
    const float* pen_bd = (const float*)d_in[39];
    const float* pen_Wl = (const float*)d_in[40];
    const float* pen_bl = (const float*)d_in[41];
    const float* fc1_W  = (const float*)d_in[42];
    const float* fc1_b  = (const float*)d_in[43];
    const float* fc2_W  = (const float*)d_in[44];
    const float* fc2_b  = (const float*)d_in[45];
    const int* node_id_d = (const int*)d_in[46];
    const int* node_id_l = (const int*)d_in[47];
    const int* ei  = (const int*)d_in[48];   // [2,E]: row0=s (disease), row1=t (lncrna)
    const int* eli = (const int*)d_in[49];   // [2,EL]
    const int* s_idx = ei;
    const int* t_idx = ei + E;
    const int* el0 = eli;
    const int* el1 = eli + EL;

    // ---- workspace layout ----
    float* ws = (float*)d_ws;
    float* xd_a = ws;                 ws += N * 128;
    float* xl_a = ws;                 ws += N * 128;
    float* xd_b = ws;                 ws += N * 128;
    float* xl_b = ws;                 ws += N * 128;
    float* hs   = ws;                 ws += N * 512;   // reused later as ef [EL,64]
    float* agg  = ws;                 ws += N * 512;   // reused later as h1 [EL,64]
    float* es   = ws;                 ws += N * NHEAD;
    float* ed   = ws;                 ws += N * NHEAD;
    unsigned* menc = (unsigned*)ws;   ws += N * NHEAD;
    float* den  = ws;                 ws += N * NHEAD;
    float* ebuf = ws;                 ws += E * NHEAD;
    float* weff_s = ws;               ws += 128 * NHEAD;
    float* weff_d = ws;               ws += 128 * NHEAD;
    __bf16* Apack = (__bf16*)ws;      ws += (N * 416) / 2 + 64;  // N x 416 bf16
    __bf16* Bpack = (__bf16*)ws;      ws += (416 * 512) / 2 + 64;

    // ---- input linears (WMMA GEMM + bias + embedding gather) ----
    gemm_full(x_d, Wd_lin, bd_lin, emb_d, node_id_d, xd_a, (int)N, 412, 128, 0,
              Apack, Bpack, stream);
    gemm_full(x_l, Wl_lin, bl_lin, emb_l, node_id_l, xl_a, (int)N, 240, 128, 0,
              Apack, Bpack, stream);

    // ---- 3 GAT layers, two directions each ----
    float* cur_xd = xd_a; float* cur_xl = xl_a;
    float* nxt_xd = xd_b; float* nxt_xl = xl_b;
    for (int L = 0; L < 3; ++L) {
        int Kin = 128;
        int C = (L < 2) ? 128 : 32;
        // d2l: src = disease via s, dst = lncrna via t -> new_l
        run_gat(cur_xd, cur_xl,
                g[L][0][0], g[L][0][1], g[L][0][2], g[L][0][3], g[L][0][4],
                s_idx, t_idx, Kin, C, nxt_xl,
                hs, agg, es, ed, menc, den, ebuf, weff_s, weff_d,
                Apack, Bpack, N, E, stream);
        // l2d: src = lncrna via t, dst = disease via s -> new_d
        run_gat(cur_xl, cur_xd,
                g[L][1][0], g[L][1][1], g[L][1][2], g[L][1][3], g[L][1][4],
                t_idx, s_idx, Kin, C, nxt_xd,
                hs, agg, es, ed, menc, den, ebuf, weff_s, weff_d,
                Apack, Bpack, N, E, stream);
        float* tmp;
        tmp = cur_xd; cur_xd = nxt_xd; nxt_xd = tmp;
        tmp = cur_xl; cur_xl = nxt_xl; nxt_xl = tmp;
    }

    // ---- penalty scaling (C = 32 now) ----
    penalty_scale<<<nblk(N, 256), 256, 0, stream>>>(cur_xd, pen_Wd, pen_bd, N);
    penalty_scale<<<nblk(N, 256), 256, 0, stream>>>(cur_xl, pen_Wl, pen_bl, N);

    // ---- link classifier on supervision edges ----
    float* ef = hs;   // [EL, 64]
    float* h1 = agg;  // [EL, 64]
    gather_ef<<<nblk(EL * 64, 256), 256, 0, stream>>>(cur_xd, cur_xl, el0, el1, ef, EL);
    gemm_full(ef, fc1_W, fc1_b, nullptr, nullptr, h1, (int)EL, 64, 64, 1,
              Apack, Bpack, stream);
    fc2_dot<<<nblk(EL, 256), 256, 0, stream>>>(h1, fc2_W, fc2_b, (float*)d_out, EL);
}
